// RWKV_Tmix_x070_85804856640276
// MI455X (gfx1250) — compile-verified, matrix operand order checked by reading, not run on hardware
//
#include <hip/hip_runtime.h>

// ---------------------------------------------------------------------------
// Problem constants (match the reference)
// ---------------------------------------------------------------------------
#define B_ 2
#define T_ 2048
#define C_ 1024
#define H_ 16
#define N_ 64
#define DW_ 64
#define DG_ 160
#define BT_ (B_ * T_)           // 4096
#define BTC_ ((size_t)BT_ * C_) // 4194304
#define EPS_LN_ 0.00064f

#ifndef __has_builtin
#define __has_builtin(x) 0
#endif
#if __has_builtin(__builtin_amdgcn_global_load_async_to_lds_b128) && \
    __has_builtin(__builtin_amdgcn_s_wait_asynccnt)
#define USE_ASYNC_SCAN 1
#else
#define USE_ASYNC_SCAN 0
#endif

typedef __attribute__((ext_vector_type(16))) __bf16 v16bf;
typedef __attribute__((ext_vector_type(8))) float v8f;
typedef __attribute__((ext_vector_type(8))) unsigned int v8u;

// float -> bf16 bits, round-to-nearest-even
static __device__ __forceinline__ unsigned short f2bf(float f) {
  unsigned int u = __float_as_uint(f);
  u += 0x7FFFu + ((u >> 16) & 1u);
  return (unsigned short)(u >> 16);
}

static __device__ __forceinline__ float sigmoidf_(float x) {
  return 1.0f / (1.0f + __expf(-x));
}

// ---------------------------------------------------------------------------
// 16-bit WMMA fragment loader (layout per CDNA5 ISA, 16-bit A
// 16x32): lane half = lane>>4, row = lane&15; VGPR p<4 holds K pair
// (half*8 + 2p), p>=4 holds (16 + half*8 + 2(p-4)). B uses the same pairing
// with row = N since Bt is stored N-major. Pairs p=0..3 and p=4..7 are each
// 16 contiguous bytes -> compiler merges into two global_load_b128.
// ---------------------------------------------------------------------------
static __device__ __forceinline__ v16bf load_frag16(
    const unsigned short* __restrict__ rowbase, int k0, int half) {
  v8u r;
#pragma unroll
  for (int p = 0; p < 8; ++p) {
    int kk = k0 + ((p < 4) ? (half * 8 + 2 * p) : (16 + half * 8 + 2 * (p - 4)));
    r[p] = *reinterpret_cast<const unsigned int*>(rowbase + kk);
  }
  return __builtin_bit_cast(v16bf, r);
}

// ---------------------------------------------------------------------------
// GEMM: C[M,N] (f32) = A[M,K] (bf16, row-major) x Bt[N,K] (bf16, N-major).
// One wave computes a 32 x (16*NT) tile: 2 A fragments reused across NT B
// fragments -> 2*NT independent v_wmma_f32_16x16x32_bf16 chains per K-step.
// ---------------------------------------------------------------------------
template <int NT>
__global__ void __launch_bounds__(256) gemm_bf16_wmma(
    const unsigned short* __restrict__ A,
    const unsigned short* __restrict__ Bt,
    float* __restrict__ Cm, int M, int N, int K) {
  int wave = (int)((blockIdx.x * blockDim.x + threadIdx.x) >> 5);
  int lane = threadIdx.x & 31;
  int tilesN = N / (16 * NT);
  int totalWaves = (M / 32) * tilesN;
  if (wave >= totalWaves) return;

  int m0 = (wave / tilesN) * 32;
  int n0 = (wave % tilesN) * (16 * NT);
  int half = lane >> 4;
  int mr = lane & 15;

  const unsigned short* arow0 = A + (size_t)(m0 + mr) * K;
  const unsigned short* arow1 = A + (size_t)(m0 + 16 + mr) * K;
  const unsigned short* brow[NT];
#pragma unroll
  for (int t = 0; t < NT; ++t)
    brow[t] = Bt + (size_t)(n0 + t * 16 + mr) * K;

  v8f acc0[NT] = {};
  v8f acc1[NT] = {};
  for (int k0 = 0; k0 < K; k0 += 32) {
    if (k0 + 32 < K) {
      __builtin_prefetch(arow0 + k0 + 32, 0, 1);
      __builtin_prefetch(arow1 + k0 + 32, 0, 1);
    }
    v16bf a0f = load_frag16(arow0, k0, half);
    v16bf a1f = load_frag16(arow1, k0, half);
#pragma unroll
    for (int t = 0; t < NT; ++t) {
      v16bf bf = load_frag16(brow[t], k0, half);
      acc0[t] = __builtin_amdgcn_wmma_f32_16x16x32_bf16(
          false, a0f, false, bf, (short)0, acc0[t], false, false);
      acc1[t] = __builtin_amdgcn_wmma_f32_16x16x32_bf16(
          false, a1f, false, bf, (short)0, acc1[t], false, false);
    }
  }
  // D layout: lane gives n = lane&15; VGPR r gives m = r + 8*(lane>>4)
#pragma unroll
  for (int t = 0; t < NT; ++t) {
#pragma unroll
    for (int r = 0; r < 8; ++r) {
      int n = n0 + t * 16 + mr;
      Cm[(size_t)(m0 + r + half * 8) * N + n] = acc0[t][r];
      Cm[(size_t)(m0 + 16 + r + half * 8) * N + n] = acc1[t][r];
    }
  }
}

// ---------------------------------------------------------------------------
// Weight transpose + f32->bf16: W (R,Ccol) row-major -> Wt (Ccol,R) row-major.
// ---------------------------------------------------------------------------
__global__ void transpose_to_bf16(const float* __restrict__ W,
                                  unsigned short* __restrict__ Wt,
                                  int R, int Ccol) {
  int idx = blockIdx.x * blockDim.x + threadIdx.x;
  if (idx >= R * Ccol) return;
  int c = idx / R;
  int r = idx - c * R;
  Wt[(size_t)c * R + r] = f2bf(W[(size_t)r * Ccol + c]);
}

// ---------------------------------------------------------------------------
// Token-shift + six mixes -> bf16; also passthrough copy of v_first.
// ---------------------------------------------------------------------------
__global__ void prep_kernel(
    const float* __restrict__ x, const float* __restrict__ vfirst,
    const float* __restrict__ mr, const float* __restrict__ mw,
    const float* __restrict__ mk, const float* __restrict__ mv,
    const float* __restrict__ ma, const float* __restrict__ mg,
    unsigned short* __restrict__ oxr, unsigned short* __restrict__ oxw,
    unsigned short* __restrict__ oxk, unsigned short* __restrict__ oxv,
    unsigned short* __restrict__ oxa, unsigned short* __restrict__ oxg,
    float* __restrict__ vf_out) {
  size_t idx = (size_t)blockIdx.x * blockDim.x + threadIdx.x;
  int c = (int)(idx & (C_ - 1));
  size_t tc = idx >> 10;          // b*T + t   (C_ == 1024)
  int t = (int)(tc & (T_ - 1));   // T_ power of two
  float xv = x[idx];
  float prev = (t == 0) ? 0.0f : x[idx - C_];
  float xx = prev - xv;
  oxr[idx] = f2bf(xv + xx * mr[c]);
  oxw[idx] = f2bf(xv + xx * mw[c]);
  oxk[idx] = f2bf(xv + xx * mk[c]);
  oxv[idx] = f2bf(xv + xx * mv[c]);
  oxa[idx] = f2bf(xv + xx * ma[c]);
  oxg[idx] = f2bf(xv + xx * mg[c]);
  vf_out[idx] = vfirst[idx];
}

// ---------------------------------------------------------------------------
// Elementwise activation + f32 -> bf16 (mode: 0 none, 1 tanh, 2 sigmoid)
// ---------------------------------------------------------------------------
__global__ void elem_act(const float* __restrict__ in,
                         unsigned short* __restrict__ out, int n, int mode) {
  int i = blockIdx.x * blockDim.x + threadIdx.x;
  if (i >= n) return;
  float v = in[i];
  if (mode == 1) v = tanhf(v);
  else if (mode == 2) v = sigmoidf_(v);
  out[i] = f2bf(v);
}

// ---------------------------------------------------------------------------
// Per-token post-LoRA elementwise: decay, a, v-mix, per-head kk norm, k mod.
// One 1024-thread block per token. Some out pointers alias in pointers
// elementwise (read-before-write per index) -> no __restrict__ on those.
// ---------------------------------------------------------------------------
__global__ void __launch_bounds__(1024) act2_kernel(
    const float* wpre, const float* apre, const float* vpre,
    const float* __restrict__ vraw, const float* __restrict__ vfirst,
    float* kbuf,                       // in: k raw, out: k modulated
    float* dh, float* aout, float* vout,
    float* __restrict__ kkout,
    const float* __restrict__ w0, const float* __restrict__ a0,
    const float* __restrict__ v0, const float* __restrict__ k_k,
    const float* __restrict__ k_a) {
  __shared__ float red[1024];
  int tid = threadIdx.x;
  int c = tid;
  size_t idx = (size_t)blockIdx.x * 1024 + tid;

  float wp = w0[c] + wpre[idx];
  float w = -log1pf(__expf(-wp)) - 0.5f;   // -softplus(-wp) - 0.5
  float d = __expf(w);
  float a = sigmoidf_(a0[c] + apre[idx]);
  float vm = sigmoidf_(v0[c] + vpre[idx]);
  float vr = vraw[idx];
  float v = vr + (vfirst[idx] - vr) * vm;
  float kraw = kbuf[idx];
  float kkraw = kraw * k_k[c];

  red[tid] = kkraw * kkraw;
  __syncthreads();
#pragma unroll
  for (int s = 32; s > 0; s >>= 1) {
    if ((tid & 63) < s) red[tid] += red[tid + s];
    __syncthreads();
  }
  float norm = sqrtf(red[tid & ~63]);
  float kkn = kkraw / fmaxf(norm, 1e-12f);
  float kmod = kraw * (1.0f + (a - 1.0f) * k_a[c]);

  dh[idx] = d;
  aout[idx] = a;
  vout[idx] = v;
  kkout[idx] = kkn;
  kbuf[idx] = kmod;
}

// ---------------------------------------------------------------------------
// Sequential WKV scan: one block (256 threads) per (b,h); 64x64 f32 state in
// registers (4 lanes per state row -> __shfl_xor row reductions). Next step's
// six 64-float vectors are double-buffered in LDS and fetched with
// GLOBAL_LOAD_ASYNC_TO_LDS_B128 (ASYNCcnt) when available; otherwise staged
// through registers. 2 barriers per step either way.
// ---------------------------------------------------------------------------
__global__ void __launch_bounds__(256) scan_kernel(
    const float* __restrict__ r, const float* __restrict__ dh,
    const float* __restrict__ k, const float* __restrict__ v,
    const float* __restrict__ kk, const float* __restrict__ a,
    float* __restrict__ o) {
  __shared__ float rbuf[2][64], dbuf[2][64], kbuf[2][64];
  __shared__ float vbuf[2][64], kkbuf[2][64], abuf[2][64];

  int b = blockIdx.x >> 4;   // H_ = 16
  int h = blockIdx.x & 15;
  int tid = threadIdx.x;
  int i = tid >> 2;          // state row
  int q = tid & 3;           // column quarter (16 cols)

  float S[16];
#pragma unroll
  for (int jj = 0; jj < 16; ++jj) S[jj] = 0.0f;

  size_t headoff = (size_t)h * 64;
  const float* srcs[6] = {r, dh, k, v, kk, a};

#if USE_ASYNC_SCAN
  // builtin signature (from hipcc diagnostic): (v4i AS1*, v4i AS3*, Ii, Ii)
  typedef int v4i_ __attribute__((vector_size(16)));
  typedef __attribute__((address_space(1))) v4i_ as1_v4i;
  typedef __attribute__((address_space(3))) v4i_ as3_v4i;
  // 6 arrays x 16 lanes of B128 (16 bytes = 4 floats): lanes 0..95 issue one
  // async instruction per half-wave -> 2 async ops per wave (waves 0..2).
  auto issue = [&](int t, int p) {
    if (tid < 96) {
      int arr = tid >> 4;
      int l = tid & 15;
      size_t base = ((size_t)b * T_ + t) * C_ + headoff + (size_t)l * 4;
      const float* g = srcs[arr] + base;
      float* lb;
      switch (arr) {
        case 0: lb = &rbuf[p][l * 4]; break;
        case 1: lb = &dbuf[p][l * 4]; break;
        case 2: lb = &kbuf[p][l * 4]; break;
        case 3: lb = &vbuf[p][l * 4]; break;
        case 4: lb = &kkbuf[p][l * 4]; break;
        default: lb = &abuf[p][l * 4]; break;
      }
      __builtin_amdgcn_global_load_async_to_lds_b128(
          (as1_v4i*)(void*)g, (as3_v4i*)lb, 0, 0);
    }
  };
  issue(0, 0);
#else
  float pr0 = 0.0f, pr1 = 0.0f, pr2 = 0.0f;
  {
    size_t base = ((size_t)b * T_) * C_ + headoff;
    if (tid < 64) {
      rbuf[0][tid] = r[base + tid];
      dbuf[0][tid] = dh[base + tid];
      kbuf[0][tid] = k[base + tid];
    } else if (tid < 128) {
      int n = tid - 64;
      vbuf[0][n] = v[base + n];
      kkbuf[0][n] = kk[base + n];
      abuf[0][n] = a[base + n];
    }
  }
  __syncthreads();
#endif

  for (int t = 0; t < T_; ++t) {
    size_t base = ((size_t)b * T_ + t) * C_ + headoff;
#if USE_ASYNC_SCAN
    if (t + 1 < T_) issue(t + 1, (t + 1) & 1);
    if (tid < 96) {  // wave-uniform: waves 0..2 issued the async loads
      if (t + 1 < T_) __builtin_amdgcn_s_wait_asynccnt(2);
      else            __builtin_amdgcn_s_wait_asynccnt(0);
    }
    __syncthreads();
    int pb = t & 1;
#else
    if (t + 1 < T_) {
      size_t nbase = base + C_;
      if (tid < 64) {
        pr0 = r[nbase + tid];
        pr1 = dh[nbase + tid];
        pr2 = k[nbase + tid];
      } else if (tid < 128) {
        int n = tid - 64;
        pr0 = v[nbase + n];
        pr1 = kk[nbase + n];
        pr2 = a[nbase + n];
      }
    }
    const int pb = 0;
#endif

    // sa_i = sum_j S_ij * (-kk_j)   (pre-decay S, matches reference)
    float p = 0.0f;
#pragma unroll
    for (int jj = 0; jj < 16; ++jj) p -= S[jj] * kkbuf[pb][q * 16 + jj];
    p += __shfl_xor(p, 1);
    p += __shfl_xor(p, 2);
    float sai = p;
    float vi = vbuf[pb][i];
    float yp = 0.0f;
#pragma unroll
    for (int jj = 0; jj < 16; ++jj) {
      int j = q * 16 + jj;
      float bj = kkbuf[pb][j] * abuf[pb][j];   // b_t = kk * a
      float s = S[jj] * dbuf[pb][j] + sai * bj + vi * kbuf[pb][j];
      S[jj] = s;
      yp += s * rbuf[pb][j];
    }
    yp += __shfl_xor(yp, 1);
    yp += __shfl_xor(yp, 2);
    if (q == 0) o[base + i] = yp;

    __syncthreads();  // all reads of current buffer done before it is refilled
#if !USE_ASYNC_SCAN
    if (t + 1 < T_) {
      if (tid < 64) {
        rbuf[0][tid] = pr0;
        dbuf[0][tid] = pr1;
        kbuf[0][tid] = pr2;
      } else if (tid < 128) {
        int n = tid - 64;
        vbuf[0][n] = pr0;
        kkbuf[0][n] = pr1;
        abuf[0][n] = pr2;
      }
      __syncthreads();
    }
#endif
  }
}

// ---------------------------------------------------------------------------
// Per-token GroupNorm (per head over N=64) + r*r_k*k bonus + gate -> bf16.
// ---------------------------------------------------------------------------
__global__ void __launch_bounds__(1024) gn_kernel(
    const float* __restrict__ o, const float* __restrict__ r,
    const float* __restrict__ kmod, const float* __restrict__ v,
    const float* __restrict__ g, const float* __restrict__ r_k,
    const float* __restrict__ ln_g, const float* __restrict__ ln_b,
    unsigned short* __restrict__ og) {
  __shared__ float red1[1024];
  __shared__ float red2[1024];
  int tid = threadIdx.x;
  int c = tid;
  size_t idx = (size_t)blockIdx.x * 1024 + tid;

  float ov = o[idx];
  red1[tid] = ov;
  red2[tid] = ov * ov;
  __syncthreads();
#pragma unroll
  for (int s = 32; s > 0; s >>= 1) {
    if ((tid & 63) < s) {
      red1[tid] += red1[tid + s];
      red2[tid] += red2[tid + s];
    }
    __syncthreads();
  }
  float mu = red1[tid & ~63] * (1.0f / 64.0f);
  float var = red2[tid & ~63] * (1.0f / 64.0f) - mu * mu;
  float onv = (ov - mu) * rsqrtf(var + EPS_LN_) * ln_g[c] + ln_b[c];
  __syncthreads();

  red1[tid] = r[idx] * r_k[c] * kmod[idx];  // r_k flat (H,N) == c index
  __syncthreads();
#pragma unroll
  for (int s = 32; s > 0; s >>= 1) {
    if ((tid & 63) < s) red1[tid] += red1[tid + s];
    __syncthreads();
  }
  float dot = red1[tid & ~63];
  float res = (onv + dot * v[idx]) * g[idx];
  og[idx] = f2bf(res);
}

// ---------------------------------------------------------------------------
// Host orchestration
// ---------------------------------------------------------------------------
extern "C" void kernel_launch(void* const* d_in, const int* in_sizes, int n_in,
                              void* d_out, int out_size, void* d_ws, size_t ws_size,
                              hipStream_t stream) {
  (void)in_sizes; (void)n_in; (void)out_size; (void)ws_size;

  const float* x       = (const float*)d_in[0];
  const float* v_first = (const float*)d_in[1];
  const float* x_r = (const float*)d_in[2];
  const float* x_w = (const float*)d_in[3];
  const float* x_k = (const float*)d_in[4];
  const float* x_v = (const float*)d_in[5];
  const float* x_a = (const float*)d_in[6];
  const float* x_g = (const float*)d_in[7];
  const float* w0 = (const float*)d_in[8];
  const float* w1 = (const float*)d_in[9];
  const float* w2 = (const float*)d_in[10];
  const float* a0 = (const float*)d_in[11];
  const float* a1 = (const float*)d_in[12];
  const float* a2 = (const float*)d_in[13];
  const float* v0 = (const float*)d_in[14];
  const float* v1 = (const float*)d_in[15];
  const float* v2 = (const float*)d_in[16];
  const float* g1 = (const float*)d_in[17];
  const float* g2 = (const float*)d_in[18];
  const float* k_k = (const float*)d_in[19];
  const float* k_a = (const float*)d_in[20];
  const float* r_k = (const float*)d_in[21];
  const float* Wr = (const float*)d_in[22];
  const float* Wk = (const float*)d_in[23];
  const float* Wv = (const float*)d_in[24];
  const float* Wo = (const float*)d_in[25];
  const float* ln_g = (const float*)d_in[26];
  const float* ln_b = (const float*)d_in[27];

  float* out_f = (float*)d_out;           // [BTC] main output
  float* vf_out = out_f + BTC_;           // [BTC] v_first passthrough

  // Bump allocator over workspace (256B aligned)
  char* wsb = (char*)d_ws;
  size_t off = 0;
  auto alloc = [&](size_t bytes) -> void* {
    off = (off + 255) & ~(size_t)255;
    void* p = wsb + off;
    off += bytes;
    return p;
  };

  // bf16 mixed activations
  unsigned short* bxr = (unsigned short*)alloc(BTC_ * 2);
  unsigned short* bxw = (unsigned short*)alloc(BTC_ * 2);
  unsigned short* bxk = (unsigned short*)alloc(BTC_ * 2);
  unsigned short* bxv = (unsigned short*)alloc(BTC_ * 2);
  unsigned short* bxa = (unsigned short*)alloc(BTC_ * 2);
  unsigned short* bxg = (unsigned short*)alloc(BTC_ * 2);
  // bf16 transposed weights
  unsigned short* WrT = (unsigned short*)alloc((size_t)C_ * C_ * 2);
  unsigned short* WkT = (unsigned short*)alloc((size_t)C_ * C_ * 2);
  unsigned short* WvT = (unsigned short*)alloc((size_t)C_ * C_ * 2);
  unsigned short* WoT = (unsigned short*)alloc((size_t)C_ * C_ * 2);
  unsigned short* w1T = (unsigned short*)alloc((size_t)C_ * DW_ * 2);
  unsigned short* w2T = (unsigned short*)alloc((size_t)C_ * DW_ * 2);
  unsigned short* a1T = (unsigned short*)alloc((size_t)C_ * DW_ * 2);
  unsigned short* a2T = (unsigned short*)alloc((size_t)C_ * DW_ * 2);
  unsigned short* v1T = (unsigned short*)alloc((size_t)C_ * DW_ * 2);
  unsigned short* v2T = (unsigned short*)alloc((size_t)C_ * DW_ * 2);
  unsigned short* g1T = (unsigned short*)alloc((size_t)C_ * DG_ * 2);
  unsigned short* g2T = (unsigned short*)alloc((size_t)C_ * DG_ * 2);
  // f32 intermediates
  float* r_f    = (float*)alloc(BTC_ * 4);
  float* k_f    = (float*)alloc(BTC_ * 4);   // k raw, then k modulated
  float* vraw_f = (float*)alloc(BTC_ * 4);
  float* wpre_f = (float*)alloc(BTC_ * 4);   // then decay d
  float* apre_f = (float*)alloc(BTC_ * 4);   // then a
  float* vpre_f = (float*)alloc(BTC_ * 4);   // then v (mixed)
  float* g_f    = (float*)alloc(BTC_ * 4);
  float* kk_f   = (float*)alloc(BTC_ * 4);
  float* o_f    = (float*)alloc(BTC_ * 4);
  float* hw_f = (float*)alloc((size_t)BT_ * DW_ * 4);
  float* ha_f = (float*)alloc((size_t)BT_ * DW_ * 4);
  float* hv_f = (float*)alloc((size_t)BT_ * DW_ * 4);
  float* hg_f = (float*)alloc((size_t)BT_ * DG_ * 4);
  unsigned short* thw = (unsigned short*)alloc((size_t)BT_ * DW_ * 2);
  unsigned short* tha = (unsigned short*)alloc((size_t)BT_ * DW_ * 2);
  unsigned short* thv = (unsigned short*)alloc((size_t)BT_ * DW_ * 2);
  unsigned short* thg = (unsigned short*)alloc((size_t)BT_ * DG_ * 2);
  unsigned short* og_b = (unsigned short*)alloc(BTC_ * 2);

  // --- 1) token shift + mixes + v_first passthrough ---
  prep_kernel<<<(unsigned)(BTC_ / 256), 256, 0, stream>>>(
      x, v_first, x_r, x_w, x_k, x_v, x_a, x_g,
      bxr, bxw, bxk, bxv, bxa, bxg, vf_out);

  // --- 2) weight transposes to bf16 ---
  auto tr = [&](const float* W, unsigned short* Wt, int R, int Ccol) {
    int n = R * Ccol;
    transpose_to_bf16<<<(n + 255) / 256, 256, 0, stream>>>(W, Wt, R, Ccol);
  };
  tr(Wr, WrT, C_, C_);
  tr(Wk, WkT, C_, C_);
  tr(Wv, WvT, C_, C_);
  tr(Wo, WoT, C_, C_);
  tr(w1, w1T, C_, DW_);   // (1024,64) -> (64,1024)
  tr(w2, w2T, DW_, C_);   // (64,1024) -> (1024,64)
  tr(a1, a1T, C_, DW_);
  tr(a2, a2T, DW_, C_);
  tr(v1, v1T, C_, DW_);
  tr(v2, v2T, DW_, C_);
  tr(g1, g1T, C_, DG_);   // (1024,160) -> (160,1024)
  tr(g2, g2T, DG_, C_);   // (160,1024) -> (1024,160)

  auto gemm4 = [&](const unsigned short* A, const unsigned short* Bt, float* Cm,
                   int M, int N, int K) {
    int waves = (M / 32) * (N / 64);
    int blocks = (waves + 7) / 8;
    gemm_bf16_wmma<4><<<blocks, 256, 0, stream>>>(A, Bt, Cm, M, N, K);
  };
  auto gemm1 = [&](const unsigned short* A, const unsigned short* Bt, float* Cm,
                   int M, int N, int K) {
    int waves = (M / 32) * (N / 16);
    int blocks = (waves + 7) / 8;
    gemm_bf16_wmma<1><<<blocks, 256, 0, stream>>>(A, Bt, Cm, M, N, K);
  };

  // --- 3) big projections ---
  gemm4(bxr, WrT, r_f, BT_, C_, C_);
  gemm4(bxk, WkT, k_f, BT_, C_, C_);
  gemm4(bxv, WvT, vraw_f, BT_, C_, C_);

  // --- 4) LoRA first halves ---
  gemm4(bxw, w1T, hw_f, BT_, DW_, C_);
  gemm4(bxa, a1T, ha_f, BT_, DW_, C_);
  gemm4(bxv, v1T, hv_f, BT_, DW_, C_);
  gemm1(bxg, g1T, hg_f, BT_, DG_, C_);

  // --- 5) inner activations -> bf16 ---
  int nw = BT_ * DW_, ng = BT_ * DG_;
  elem_act<<<(nw + 255) / 256, 256, 0, stream>>>(hw_f, thw, nw, 1); // tanh
  elem_act<<<(nw + 255) / 256, 256, 0, stream>>>(ha_f, tha, nw, 0);
  elem_act<<<(nw + 255) / 256, 256, 0, stream>>>(hv_f, thv, nw, 0);
  elem_act<<<(ng + 255) / 256, 256, 0, stream>>>(hg_f, thg, ng, 2); // sigmoid

  // --- 6) LoRA second halves ---
  gemm4(thw, w2T, wpre_f, BT_, C_, DW_);
  gemm4(tha, a2T, apre_f, BT_, C_, DW_);
  gemm4(thv, v2T, vpre_f, BT_, C_, DW_);
  gemm4(thg, g2T, g_f, BT_, C_, DG_);

  // --- 7) per-token elementwise (decay, a, v-mix, kk-norm, k-mod) ---
  act2_kernel<<<BT_, 1024, 0, stream>>>(
      wpre_f, apre_f, vpre_f, vraw_f, v_first, k_f,
      wpre_f /*d out*/, apre_f /*a out*/, vpre_f /*v out*/, kk_f,
      w0, a0, v0, k_k, k_a);

  // --- 8) sequential WKV scan over T (32 independent (b,h) scans) ---
  scan_kernel<<<B_ * H_, 256, 0, stream>>>(
      r_f, wpre_f /*decay*/, k_f /*kmod*/, vpre_f /*v*/, kk_f, apre_f /*a*/, o_f);

  // --- 9) GroupNorm + bonus + gate -> bf16 ---
  gn_kernel<<<BT_, 1024, 0, stream>>>(
      o_f, r_f, k_f, vpre_f, g_f, r_k, ln_g, ln_b, og_b);

  // --- 10) output projection ---
  gemm4(og_b, WoT, out_f, BT_, C_, C_);
}
